// Top2MoE_37769942401311
// MI455X (gfx1250) — matrix operand order repaired and checked by reading, not compiled, not run on hardware
//
#include <hip/hip_runtime.h>
#include <hip/hip_bf16.h>
#include <math.h>

// ---------------- problem constants (from reference) ----------------
#define T_TOK   8192
#define D_DIM   1024
#define F_DIM   4096
#define E_EXP   8
#define CAPCTY  2560              // int(1.25 * 2T / E)
#define PAD_SL  (E_EXP * CAPCTY)  // 20480
#define TWO_T   (2 * T_TOK)

// ---------------- WMMA vector types (gfx1250) ----------------
typedef __bf16 bf16x8  __attribute__((ext_vector_type(8)));
typedef __bf16 v16bf   __attribute__((ext_vector_type(16)));
typedef float  v8f     __attribute__((ext_vector_type(8)));
typedef int    v4i     __attribute__((ext_vector_type(4)));

// ---------------- CDNA5 async global->LDS path -----------------------
#if __has_builtin(__builtin_amdgcn_global_load_async_to_lds_b128)
#define HAVE_ASYNC_LDS 1
#define AS_GLOBAL __attribute__((address_space(1)))
#define AS_LOCAL  __attribute__((address_space(3)))
// param1: v4i addrspace(1)* (global), param2: v4i addrspace(3)* (LDS).
// Cast through integers: global AS == flat address; LDS offset == flat[31:0].
#define ASYNC_CP16(gsrc, ldst)                                             \
  __builtin_amdgcn_global_load_async_to_lds_b128(                          \
      (AS_GLOBAL v4i*)(unsigned long long)(gsrc),                          \
      (AS_LOCAL v4i*)(unsigned)(unsigned long long)(ldst), 0, 0)
#else
#define HAVE_ASYNC_LDS 0
#endif

__device__ __forceinline__ void wait_async_all() {
#if HAVE_ASYNC_LDS
#if __has_builtin(__builtin_amdgcn_s_wait_asynccnt)
  __builtin_amdgcn_s_wait_asynccnt(0);
#else
  asm volatile("s_wait_asynccnt 0x0" ::: "memory");
#endif
#endif
}

// =====================================================================
// 1) init: slot->assignment table to -1
// =====================================================================
__global__ __launch_bounds__(256) void init_kernel(int* __restrict__ slotTok) {
  int i = blockIdx.x * 256 + threadIdx.x;
  if (i < PAD_SL) slotTok[i] = -1;
}

// =====================================================================
// 2) route: logits = x @ Wr, softmax, top2, gates; store probs
// one block per token
// =====================================================================
__global__ __launch_bounds__(256) void route_kernel(
    const float* __restrict__ x, const float* __restrict__ Wr,
    float* __restrict__ probs,   // [T, E]
    float* __restrict__ gates,   // [2T]
    int*   __restrict__ flatE)   // [2T]
{
  __shared__ float red[256 * E_EXP];
  const int t = blockIdx.x, tid = threadIdx.x;
  const float* xr = x + (size_t)t * D_DIM;

  float acc[E_EXP];
#pragma unroll
  for (int e = 0; e < E_EXP; ++e) acc[e] = 0.f;

  for (int d = tid; d < D_DIM; d += 256) {
    const float xv = xr[d];
    const float* wr = Wr + (size_t)d * E_EXP;
#pragma unroll
    for (int e = 0; e < E_EXP; ++e) acc[e] += xv * wr[e];
  }
#pragma unroll
  for (int e = 0; e < E_EXP; ++e) red[tid * E_EXP + e] = acc[e];
  __syncthreads();
  for (int s = 128; s > 0; s >>= 1) {
    if (tid < s)
#pragma unroll
      for (int e = 0; e < E_EXP; ++e)
        red[tid * E_EXP + e] += red[(tid + s) * E_EXP + e];
    __syncthreads();
  }
  if (tid == 0) {
    float l[E_EXP], p[E_EXP];
    float mx = red[0];
#pragma unroll
    for (int e = 0; e < E_EXP; ++e) { l[e] = red[e]; mx = fmaxf(mx, l[e]); }
    float sum = 0.f;
#pragma unroll
    for (int e = 0; e < E_EXP; ++e) { p[e] = expf(l[e] - mx); sum += p[e]; }
#pragma unroll
    for (int e = 0; e < E_EXP; ++e) { p[e] /= sum; probs[(size_t)t * E_EXP + e] = p[e]; }
    // top-2, lower index wins ties (strict >)
    int i1 = 0;
#pragma unroll
    for (int e = 1; e < E_EXP; ++e) if (p[e] > p[i1]) i1 = e;
    int i2 = (i1 == 0) ? 1 : 0;
#pragma unroll
    for (int e = 0; e < E_EXP; ++e) if (e != i1 && p[e] > p[i2]) i2 = e;
    const float denom = fmaxf(p[i1] + p[i2], 1e-9f);
    flatE[2 * t]     = i1;  gates[2 * t]     = p[i1] / denom;
    flatE[2 * t + 1] = i2;  gates[2 * t + 1] = p[i2] / denom;
  }
}

// =====================================================================
// 3) rank: ordered scan over 2T assignments per expert (one block/expert)
// =====================================================================
__global__ __launch_bounds__(256) void rank_kernel(
    const int* __restrict__ flatE,
    int*   __restrict__ slot,     // [2T], -1 if dropped
    int*   __restrict__ slotTok,  // [PAD], flat assignment idx per slot
    float* __restrict__ loadc)    // [E]
{
  const int e = blockIdx.x, tid = threadIdx.x;
  const int CHUNK = TWO_T / 256;  // 64
  __shared__ int cnt[256];
  __shared__ int excl[257];

  const int base = tid * CHUNK;
  int c = 0;
  for (int j = 0; j < CHUNK; ++j) c += (flatE[base + j] == e) ? 1 : 0;
  cnt[tid] = c;
  __syncthreads();
  if (tid == 0) {
    int run = 0;
    for (int i = 0; i < 256; ++i) { excl[i] = run; run += cnt[i]; }
    excl[256] = run;
  }
  __syncthreads();
  int running = excl[tid];
  for (int j = 0; j < CHUNK; ++j) {
    const int i = base + j;
    if (flatE[i] == e) {
      const int r = running++;
      if (r < CAPCTY) { slot[i] = e * CAPCTY + r; slotTok[e * CAPCTY + r] = i; }
      else            { slot[i] = -1; }
    }
  }
  if (tid == 0) loadc[e] = (float)excl[256];
}

// =====================================================================
// 4) dispatch: scatter x rows into bf16 capacity buffer (one block/slot)
// =====================================================================
__global__ __launch_bounds__(256) void dispatch_kernel(
    const float* __restrict__ x, const int* __restrict__ slotTok,
    __bf16* __restrict__ Xb)     // [PAD, D]
{
  const int s = blockIdx.x;
  const int i = slotTok[s];
  __bf16* dst = Xb + (size_t)s * D_DIM;
  if (i >= 0) {
    const float* src = x + (size_t)(i >> 1) * D_DIM;
    for (int d = threadIdx.x; d < D_DIM; d += 256) dst[d] = (__bf16)src[d];
  } else {
    for (int d = threadIdx.x; d < D_DIM; d += 256) dst[d] = (__bf16)0.f;
  }
}

// =====================================================================
// 5) transpose + f32->bf16 convert (per expert): in [R,C] -> out [C,R]
// =====================================================================
__global__ __launch_bounds__(256) void transpose_bf16_kernel(
    const float* __restrict__ in, __bf16* __restrict__ out, int R, int C)
{
  __shared__ float tile[32][33];
  const int e = blockIdx.z;
  const float* ine = in + (size_t)e * R * C;
  __bf16* oute = out + (size_t)e * R * C;
  const int c0 = blockIdx.x * 32, r0 = blockIdx.y * 32;
  const int tx = threadIdx.x & 31, ty = threadIdx.x >> 5;
#pragma unroll
  for (int i = ty; i < 32; i += 8)
    tile[i][tx] = ine[(size_t)(r0 + i) * C + (c0 + tx)];
  __syncthreads();
#pragma unroll
  for (int i = ty; i < 32; i += 8)
    oute[(size_t)(c0 + i) * R + (r0 + tx)] = (__bf16)tile[tx][i];
}

// =====================================================================
// 6) grouped GEMM via v_wmma_f32_16x16x32_bf16, async-LDS double buffer
//    C[e] = act(A[e][M,K] @ Bt[e][N,K]^T + bias[e][N])
//    128x128 block tile, 256 thr = 8 waves (2 x 4), wave = 64x32 (4x2 frags)
// =====================================================================
#define TM 128
#define TN 128
#define TK 32
#define LDS_S 40   // bf16 stride: 80B rows -> conflict-free, 16B aligned

template <int M, int N, int K, bool GELU_ACT, bool OUT_BF16>
__global__ __launch_bounds__(256) void moe_gemm_kernel(
    const __bf16* __restrict__ Aall, const __bf16* __restrict__ Btall,
    const float* __restrict__ biasAll, void* __restrict__ Call)
{
  const int e  = blockIdx.z;
  const __bf16* A  = Aall  + (size_t)e * M * K;
  const __bf16* Bt = Btall + (size_t)e * N * K;
  const float* bias = biasAll + (size_t)e * N;
  const int m0 = blockIdx.x * TM, n0 = blockIdx.y * TN;

  __shared__ __bf16 Asl[2][TM * LDS_S];
  __shared__ __bf16 Bsl[2][TN * LDS_S];

  const int tid  = threadIdx.x;
  const int lane = tid & 31, wv = tid >> 5;
  const int wm = wv & 1, wn = wv >> 1;       // 2 x 4 wave grid
  const int kh  = lane >> 4;                 // lane half (0/1)
  const int l15 = lane & 15;

  // per-thread staging coordinates (compile-time K folds the addressing)
  const int rA0 = tid >> 2, cA0 = (tid & 3) * 8;         // chunk 0
  const int rA1 = (tid + 256) >> 2, cA1 = (tid & 3) * 8; // chunk 1
  const __bf16* gA0 = A  + (size_t)(m0 + rA0) * K + cA0;
  const __bf16* gA1 = A  + (size_t)(m0 + rA1) * K + cA1;
  const __bf16* gB0 = Bt + (size_t)(n0 + rA0) * K + cA0;
  const __bf16* gB1 = Bt + (size_t)(n0 + rA1) * K + cA1;

  v8f acc[4][2];
#pragma unroll
  for (int i = 0; i < 4; ++i)
#pragma unroll
    for (int j = 0; j < 2; ++j) acc[i][j] = (v8f)0.f;

  // stage one 128x32 A tile + 128x32 B tile into LDS buffer `buf`
  auto stage = [&](int buf, int k0) {
#if HAVE_ASYNC_LDS
    ASYNC_CP16(gA0 + k0, &Asl[buf][rA0 * LDS_S + cA0]);
    ASYNC_CP16(gA1 + k0, &Asl[buf][rA1 * LDS_S + cA1]);
    ASYNC_CP16(gB0 + k0, &Bsl[buf][rA0 * LDS_S + cA0]);
    ASYNC_CP16(gB1 + k0, &Bsl[buf][rA1 * LDS_S + cA1]);
#else
    *(bf16x8*)&Asl[buf][rA0 * LDS_S + cA0] = *(const bf16x8*)(gA0 + k0);
    *(bf16x8*)&Asl[buf][rA1 * LDS_S + cA1] = *(const bf16x8*)(gA1 + k0);
    *(bf16x8*)&Bsl[buf][rA0 * LDS_S + cA0] = *(const bf16x8*)(gB0 + k0);
    *(bf16x8*)&Bsl[buf][rA1 * LDS_S + cA1] = *(const bf16x8*)(gB1 + k0);
#endif
  };

  stage(0, 0);
  wait_async_all();
  __syncthreads();

  int cur = 0;
  for (int k0 = 0; k0 < K; k0 += TK, cur ^= 1) {
    if (k0 + TK < K) stage(cur ^ 1, k0 + TK);  // prefetch next tile (async DMA)

    // ISA 7.12.2 bf16 16x32 A layout: lane holds row (l&15),
    // K = kh*8..+7 (v0..3) and 16+kh*8..+7 (v4..7)
    v16bf afrag[4], bfrag[2];
#pragma unroll
    for (int i = 0; i < 4; ++i) {
      const int row = wm * 64 + i * 16 + l15;
      bf16x8 lo = *(const bf16x8*)&Asl[cur][row * LDS_S + kh * 8];
      bf16x8 hi = *(const bf16x8*)&Asl[cur][row * LDS_S + 16 + kh * 8];
      afrag[i] = __builtin_shufflevector(lo, hi, 0, 1, 2, 3, 4, 5, 6, 7,
                                         8, 9, 10, 11, 12, 13, 14, 15);
    }
    // B 32x16: lane holds col (l&15), K = kh*8..+7 (v0..3) and 16+kh*8..+7 (v4..7)
#pragma unroll
    for (int j = 0; j < 2; ++j) {
      const int col = wn * 32 + j * 16 + l15;
      bf16x8 lo = *(const bf16x8*)&Bsl[cur][col * LDS_S + kh * 8];
      bf16x8 hi = *(const bf16x8*)&Bsl[cur][col * LDS_S + 16 + kh * 8];
      bfrag[j] = __builtin_shufflevector(lo, hi, 0, 1, 2, 3, 4, 5, 6, 7,
                                         8, 9, 10, 11, 12, 13, 14, 15);
    }
#pragma unroll
    for (int i = 0; i < 4; ++i)
#pragma unroll
      for (int j = 0; j < 2; ++j)
        acc[i][j] = __builtin_amdgcn_wmma_f32_16x16x32_bf16(
            false, afrag[i], false, bfrag[j], (short)0, acc[i][j],
            false, false);

    wait_async_all();   // next tile resident in other buffer
    __syncthreads();    // everyone done reading `cur` before it is rewritten
  }

  // epilogue: bias + (GELU) + store; C/D layout: lane 0-15 N=lane M=v,
  // lane 16-31 N=lane-16 M=v+8
#pragma unroll
  for (int i = 0; i < 4; ++i) {
#pragma unroll
    for (int j = 0; j < 2; ++j) {
      const int col = n0 + wn * 32 + j * 16 + l15;
      const float bv = bias[col];
#pragma unroll
      for (int v = 0; v < 8; ++v) {
        const int row = m0 + wm * 64 + i * 16 + kh * 8 + v;
        float val = acc[i][j][v] + bv;
        if (GELU_ACT) val = 0.5f * val * (1.0f + erff(val * 0.70710678118f));
        const size_t off = (size_t)e * M * N + (size_t)row * N + col;
        if (OUT_BF16) ((__bf16*)Call)[off] = (__bf16)val;
        else          ((float*)Call)[off]  = val;
      }
    }
  }
}

// =====================================================================
// 7) combine: out[t] = sum(gate * y[slot]) or passthrough x
// =====================================================================
__global__ __launch_bounds__(256) void combine_kernel(
    const float* __restrict__ x, const float* __restrict__ Yf,
    const float* __restrict__ gates, const int* __restrict__ slot,
    float* __restrict__ out)
{
  const int idx = blockIdx.x * 256 + threadIdx.x;  // T*D total
  const int t = idx >> 10, d = idx & (D_DIM - 1);
  const int s0 = slot[2 * t], s1 = slot[2 * t + 1];
  float r = 0.f;
  bool any = false;
  if (s0 >= 0) { r += gates[2 * t]     * Yf[(size_t)s0 * D_DIM + d]; any = true; }
  if (s1 >= 0) { r += gates[2 * t + 1] * Yf[(size_t)s1 * D_DIM + d]; any = true; }
  out[idx] = any ? r : x[idx];
}

// =====================================================================
// 8) aux loss: deterministic single-block reduction
// =====================================================================
__global__ __launch_bounds__(256) void aux_kernel(
    const float* __restrict__ probs, const float* __restrict__ loadc,
    float* __restrict__ outAux)
{
  __shared__ float red[256];
  __shared__ float imp[E_EXP];
  const int tid = threadIdx.x;
  for (int e = 0; e < E_EXP; ++e) {
    float p = 0.f;
    for (int t = tid; t < T_TOK; t += 256) p += probs[(size_t)t * E_EXP + e];
    red[tid] = p;
    __syncthreads();
    for (int s = 128; s > 0; s >>= 1) {
      if (tid < s) red[tid] += red[tid + s];
      __syncthreads();
    }
    if (tid == 0) imp[e] = red[0] / (float)T_TOK;
    __syncthreads();
  }
  if (tid == 0) {
    float ls = 0.f;
    for (int e = 0; e < E_EXP; ++e) ls += loadc[e];
    ls = fmaxf(ls, 1e-9f);
    float aux = 0.f;
    for (int e = 0; e < E_EXP; ++e) aux += imp[e] * (loadc[e] / ls);
    outAux[0] = (float)E_EXP * aux;
  }
}

// =====================================================================
// host launcher
// =====================================================================
static inline size_t align256(size_t v) { return (v + 255) & ~(size_t)255; }

extern "C" void kernel_launch(void* const* d_in, const int* in_sizes, int n_in,
                              void* d_out, int out_size, void* d_ws, size_t ws_size,
                              hipStream_t stream) {
  (void)in_sizes; (void)n_in; (void)out_size; (void)ws_size;
  const float* x  = (const float*)d_in[0];   // [T, D]
  const float* Wr = (const float*)d_in[1];   // [D, E]
  const float* W1 = (const float*)d_in[2];   // [E, D, F]
  const float* b1 = (const float*)d_in[3];   // [E, F]
  const float* W2 = (const float*)d_in[4];   // [E, F, D]
  const float* b2 = (const float*)d_in[5];   // [E, D]
  float* out = (float*)d_out;                // [T*D] + 1 aux

  // workspace layout
  char* ws = (char*)d_ws;
  size_t off = 0;
  __bf16* Xb  = (__bf16*)(ws + off); off = align256(off + (size_t)PAD_SL * D_DIM * 2);
  __bf16* Hb  = (__bf16*)(ws + off); off = align256(off + (size_t)PAD_SL * F_DIM * 2);
  float*  Yf  = (float*)(ws + off);  off = align256(off + (size_t)PAD_SL * D_DIM * 4);
  __bf16* W1t = (__bf16*)(ws + off); off = align256(off + (size_t)E_EXP * D_DIM * F_DIM * 2);
  __bf16* W2t = (__bf16*)(ws + off); off = align256(off + (size_t)E_EXP * D_DIM * F_DIM * 2);
  float*  probs = (float*)(ws + off); off = align256(off + (size_t)T_TOK * E_EXP * 4);
  float*  gates = (float*)(ws + off); off = align256(off + (size_t)TWO_T * 4);
  int*    flatE = (int*)(ws + off);   off = align256(off + (size_t)TWO_T * 4);
  int*    slot  = (int*)(ws + off);   off = align256(off + (size_t)TWO_T * 4);
  int*    slotTok = (int*)(ws + off); off = align256(off + (size_t)PAD_SL * 4);
  float*  loadc = (float*)(ws + off); off = align256(off + (size_t)E_EXP * 4);

  init_kernel<<<(PAD_SL + 255) / 256, 256, 0, stream>>>(slotTok);
  route_kernel<<<T_TOK, 256, 0, stream>>>(x, Wr, probs, gates, flatE);
  rank_kernel<<<E_EXP, 256, 0, stream>>>(flatE, slot, slotTok, loadc);
  dispatch_kernel<<<PAD_SL, 256, 0, stream>>>(x, slotTok, Xb);
  transpose_bf16_kernel<<<dim3(F_DIM / 32, D_DIM / 32, E_EXP), 256, 0, stream>>>(
      W1, W1t, D_DIM, F_DIM);
  transpose_bf16_kernel<<<dim3(D_DIM / 32, F_DIM / 32, E_EXP), 256, 0, stream>>>(
      W2, W2t, F_DIM, D_DIM);
  // GEMM1: H = gelu(Xb @ W1 + b1)   [CAP, D] x [D, F]
  moe_gemm_kernel<CAPCTY, F_DIM, D_DIM, true, true>
      <<<dim3(CAPCTY / TM, F_DIM / TN, E_EXP), 256, 0, stream>>>(
          Xb, W1t, b1, (void*)Hb);
  // GEMM2: Y = Hb @ W2 + b2          [CAP, F] x [F, D]
  moe_gemm_kernel<CAPCTY, D_DIM, F_DIM, false, false>
      <<<dim3(CAPCTY / TM, D_DIM / TN, E_EXP), 256, 0, stream>>>(
          Hb, W2t, b2, (void*)Yf);
  combine_kernel<<<(T_TOK * D_DIM) / 256, 256, 0, stream>>>(x, Yf, gates, slot, out);
  aux_kernel<<<1, 256, 0, stream>>>(probs, loadc, out + (size_t)T_TOK * D_DIM);
}
